// LlamaAttention_54425825575540
// MI455X (gfx1250) — compile-verified, matrix operand order checked by reading, not run on hardware
//
#include <hip/hip_runtime.h>
#include <hip/hip_bf16.h>
#include <math.h>

typedef __attribute__((ext_vector_type(16))) _Float16 v16h;
typedef __attribute__((ext_vector_type(8)))  _Float16 v8h;
typedef __attribute__((ext_vector_type(8)))  float    v8f;
typedef __attribute__((ext_vector_type(4)))  unsigned int v4u;
typedef __attribute__((ext_vector_type(8)))  int      v8i;
typedef __attribute__((ext_vector_type(4)))  int      v4i;

#define BB 2
#define SS 2048
#define DM 2048
#define NH 16
#define DH 128

__device__ __forceinline__ v16h cat8(v8h lo, v8h hi) {
  return __builtin_shufflevector(lo, hi, 0,1,2,3,4,5,6,7,8,9,10,11,12,13,14,15);
}

// ---------------------------------------------------------------------------
// Fragment load from row-major f16, leading dim ld (A and B NT layouts are
// identical per-lane on CDNA5: lane%16 -> row, K split in 8-elem half-runs).
// ---------------------------------------------------------------------------
__device__ __forceinline__ v16h load_frag(const _Float16* __restrict__ p,
                                          int ld, int row0, int k0, int lane) {
  const int r  = lane & 15;
  const int hf = (lane >> 4) & 1;
  const _Float16* base = p + (size_t)(row0 + r) * (size_t)ld + (size_t)(k0 + hf * 8);
  v8h lo = *reinterpret_cast<const v8h*>(base);
  v8h hi = *reinterpret_cast<const v8h*>(base + 16);
  return cat8(lo, hi);
}

__device__ __forceinline__ v8f wmma16(v16h a, v16h b, v8f c) {
  return __builtin_amdgcn_wmma_f32_16x16x32_f16(false, a, false, b, (short)0, c,
                                                false, false);
}

// ---------------------------------------------------------------------------
// Tensor Data Mover: async-load a 64-row x 32-col f16 tile (row stride 2048
// elements) from global into LDS (rows packed, 64B apart). D# per ISA 8.3/8.4:
//  g0: [1:0]=count=1, [63:32]=lds_addr, [120:64]=global_addr, [127:126]=type 2
//  g1: data_size=1(2B), tensor_dim0=2048, tensor_dim1=0xFFFF,
//      tile_dim0=32, tile_dim1=64, tensor_dim0_stride=2048
// ---------------------------------------------------------------------------
__device__ __forceinline__ void tdm_load_64x32(unsigned lds_addr,
                                               const _Float16* gptr) {
  const unsigned long long ga = (unsigned long long)(size_t)gptr;
  v4u g0 = { 1u,                                  // count=1, no gather
             lds_addr,                            // LDS byte address
             (unsigned)(ga & 0xFFFFFFFFu),        // global_addr[31:0]
             ((unsigned)((ga >> 32) & 0x01FFFFFFu)) | (2u << 30) }; // +type=2
  v8i g1 = { (int)(1u << 16),                     // data_size = 2 bytes
             (int)(2048u << 16),                  // tensor_dim0[15:0]=2048
             (int)(0xFFFFu << 16),                // tensor_dim1[15:0]=65535
             (int)(32u << 16),                    // tile_dim0 = 32
             64,                                  // tile_dim1 = 64
             2048,                                // tensor_dim0_stride[31:0]
             0, 0 };
  v4i z4 = { 0, 0, 0, 0 };
#if defined(__clang_major__) && __clang_major__ >= 23
  v8i z8 = { 0, 0, 0, 0, 0, 0, 0, 0 };
  __builtin_amdgcn_tensor_load_to_lds(g0, g1, z4, z4, z8, 0);
#else
  __builtin_amdgcn_tensor_load_to_lds(g0, g1, z4, z4, 0);
#endif
}

// ---------------------------------------------------------------------------
// fp32 -> f16 conversion
// ---------------------------------------------------------------------------
__global__ void cvt_f32_f16(const float* __restrict__ src,
                            _Float16* __restrict__ dst, size_t n) {
  const size_t i = (size_t)blockIdx.x * blockDim.x + threadIdx.x;
  if (i < n) dst[i] = (_Float16)src[i];
}

// ---------------------------------------------------------------------------
// NT GEMM with TDM double-buffered LDS staging.
// C(MxN) = A(MxK) * B(NxK)^T, K = 2048. Workgroup = 4 waves -> 64x64 C block;
// wave w computes rows m0+16w..+15 over all 64 columns (4 WMMAs / K-step).
// Wave 0 drives the TDM: two 4KB tile loads per K-step, TENSORcnt pipelined.
// mode 0: f16 row-major; mode 1: f32 row-major; mode 2: f16 (b,h,d,s).
// ---------------------------------------------------------------------------
__global__ void __launch_bounds__(128) gemm_tdm_64x64(
    const _Float16* __restrict__ A, const _Float16* __restrict__ Bm,
    _Float16* __restrict__ outh, float* __restrict__ outf,
    int M, int N, int mode) {
  __shared__ _Float16 abuf[2][64 * 32];
  __shared__ _Float16 bbuf[2][64 * 32];

  const int lane = threadIdx.x & 31;
  const int w    = threadIdx.x >> 5;
  const int m0   = blockIdx.x * 64;
  const int n0   = blockIdx.y * 64;
  const int r    = lane & 15;
  const int hf   = (lane >> 4) & 1;

  v8f acc[4] = {};
  const int nsteps = DM / 32;

  if (w == 0) {  // prologue: stage K-step 0
    tdm_load_64x32((unsigned)(size_t)&abuf[0][0], A + (size_t)m0 * DM);
    tdm_load_64x32((unsigned)(size_t)&bbuf[0][0], Bm + (size_t)n0 * DM);
  }

  for (int i = 0; i < nsteps; ++i) {
    const int cur = i & 1, nxt = cur ^ 1;
    __syncthreads();  // everyone finished reading buf[nxt] (step i-1)
    if (w == 0) {
      if (i + 1 < nsteps) {
        tdm_load_64x32((unsigned)(size_t)&abuf[nxt][0],
                       A + (size_t)m0 * DM + (size_t)(i + 1) * 32);
        tdm_load_64x32((unsigned)(size_t)&bbuf[nxt][0],
                       Bm + (size_t)n0 * DM + (size_t)(i + 1) * 32);
        __builtin_amdgcn_s_wait_tensorcnt(2);  // step-i tiles landed
      } else {
        __builtin_amdgcn_s_wait_tensorcnt(0);
      }
    }
    __syncthreads();  // buf[cur] visible to all waves

    const _Float16* ab = &abuf[cur][0];
    const _Float16* bb = &bbuf[cur][0];
    v16h af = cat8(*(const v8h*)&ab[(w * 16 + r) * 32 + hf * 8],
                   *(const v8h*)&ab[(w * 16 + r) * 32 + 16 + hf * 8]);
#pragma unroll
    for (int t = 0; t < 4; ++t) {
      v16h bf = cat8(*(const v8h*)&bb[(t * 16 + r) * 32 + hf * 8],
                     *(const v8h*)&bb[(t * 16 + r) * 32 + 16 + hf * 8]);
      acc[t] = wmma16(af, bf, acc[t]);
    }
  }

#pragma unroll
  for (int t = 0; t < 4; ++t) {
#pragma unroll
    for (int i = 0; i < 8; ++i) {
      const int m = m0 + w * 16 + i + 8 * hf;
      const int n = n0 + 16 * t + r;
      const float v = acc[t][i];
      if (mode == 0) {
        outh[(size_t)m * N + n] = (_Float16)v;
      } else if (mode == 1) {
        outf[(size_t)m * N + n] = v;
      } else {
        const int b = m / SS, s = m % SS;
        const int h = n / DH, d = n % DH;
        outh[(((size_t)b * NH + h) * DH + d) * SS + s] = (_Float16)v;
      }
    }
  }
}

// ---------------------------------------------------------------------------
// RoPE in-place on (b,s,h,d) f16 tensor; one thread per (d, d+64) pair.
// ---------------------------------------------------------------------------
__global__ void rope_kernel(_Float16* __restrict__ q,
                            const float* __restrict__ cs,
                            const float* __restrict__ sn) {
  const size_t idx = (size_t)blockIdx.x * blockDim.x + threadIdx.x;
  const int d = (int)(idx & 63);
  const int h = (int)((idx >> 6) & (NH - 1));
  const int s = (int)((idx >> 10) & (SS - 1));
  const int b = (int)(idx >> 21);
  const size_t base = (((size_t)b * SS + s) * NH + h) * DH;
  const float c  = cs[s * DH + d];
  const float si = sn[s * DH + d];
  const float x1 = (float)q[base + d];
  const float x2 = (float)q[base + d + 64];
  q[base + d]      = (_Float16)(x1 * c - x2 * si);
  q[base + d + 64] = (_Float16)(x2 * c + x1 * si);
}

// ---------------------------------------------------------------------------
// Causal flash attention: one wave per (b, h, 16-row q-tile).
// Q,K: (b,s,h,d) f16; V: (b,h,d,s) f16. Online softmax; P staged in LDS.
// ---------------------------------------------------------------------------
__global__ void __launch_bounds__(32) flash_attn(
    const _Float16* __restrict__ Q, const _Float16* __restrict__ Km,
    const _Float16* __restrict__ Vt, _Float16* __restrict__ O) {
  __shared__ _Float16 pbuf[16 * 32];

  const int lane = threadIdx.x & 31;
  const int qt = blockIdx.x & (SS / 16 - 1);
  const int h  = (blockIdx.x >> 7) & (NH - 1);
  const int b  = blockIdx.x >> 11;
  const int r  = lane & 15;
  const int hf = (lane >> 4) & 1;

  const _Float16* Qp = Q  + ((size_t)b * SS) * DM + (size_t)h * DH;
  const _Float16* Kp = Km + ((size_t)b * SS) * DM + (size_t)h * DH;
  const _Float16* Vp = Vt + ((size_t)b * NH + h) * (size_t)DH * SS;

  v16h qf[4];
#pragma unroll
  for (int j = 0; j < 4; ++j) qf[j] = load_frag(Qp, DM, qt * 16, 32 * j, lane);

  v8f acc[8] = {};
  float mrow[8], lrow[8];
#pragma unroll
  for (int i = 0; i < 8; ++i) { mrow[i] = -INFINITY; lrow[i] = 0.f; }

  const float scale = 0.08838834764831845f;   // 1/sqrt(128)
  const int nch = (qt * 16 + 15) / 32 + 1;

  for (int ch = 0; ch < nch; ++ch) {
    const int j0 = ch * 32;
    float pm[2][8];
#pragma unroll
    for (int t = 0; t < 2; ++t) {
      v8f sc = {};
#pragma unroll
      for (int j = 0; j < 4; ++j) {
        v16h kf = load_frag(Kp, DM, j0 + 16 * t, 32 * j, lane);
        sc = wmma16(qf[j], kf, sc);
      }
#pragma unroll
      for (int i = 0; i < 8; ++i) {
        const int row = qt * 16 + i + 8 * hf;
        const int col = j0 + 16 * t + r;
        pm[t][i] = (col > row) ? -INFINITY : sc[i] * scale;
      }
    }

#pragma unroll
    for (int i = 0; i < 8; ++i) {
      float mx = fmaxf(pm[0][i], pm[1][i]);
#pragma unroll
      for (int off = 1; off < 16; off <<= 1)
        mx = fmaxf(mx, __shfl_xor(mx, off, 32));
      const float mnew  = fmaxf(mrow[i], mx);
      const float alpha = __expf(mrow[i] - mnew);
      mrow[i] = mnew;
      pm[0][i] = __expf(pm[0][i] - mnew);
      pm[1][i] = __expf(pm[1][i] - mnew);
      float sum = pm[0][i] + pm[1][i];
#pragma unroll
      for (int off = 1; off < 16; off <<= 1)
        sum += __shfl_xor(sum, off, 32);
      lrow[i] = lrow[i] * alpha + sum;
#pragma unroll
      for (int nt = 0; nt < 8; ++nt) acc[nt][i] *= alpha;
    }

#pragma unroll
    for (int t = 0; t < 2; ++t)
#pragma unroll
      for (int i = 0; i < 8; ++i)
        pbuf[(i + 8 * hf) * 32 + 16 * t + r] = (_Float16)pm[t][i];
    __syncthreads();
    v16h pf = cat8(*reinterpret_cast<const v8h*>(&pbuf[r * 32 + 8 * hf]),
                   *reinterpret_cast<const v8h*>(&pbuf[r * 32 + 16 + 8 * hf]));
    __syncthreads();

#pragma unroll
    for (int nt = 0; nt < 8; ++nt) {
      v16h vf = load_frag(Vp, SS, nt * 16, j0, lane);
      acc[nt] = wmma16(pf, vf, acc[nt]);
    }
  }

#pragma unroll
  for (int nt = 0; nt < 8; ++nt) {
#pragma unroll
    for (int i = 0; i < 8; ++i) {
      const int s = qt * 16 + i + 8 * hf;
      const int d = nt * 16 + r;
      O[((size_t)b * SS + s) * DM + (size_t)h * DH + d] =
          (_Float16)(acc[nt][i] / lrow[i]);
    }
  }
}

// ---------------------------------------------------------------------------
extern "C" void kernel_launch(void* const* d_in, const int* in_sizes, int n_in,
                              void* d_out, int out_size, void* d_ws, size_t ws_size,
                              hipStream_t stream) {
  (void)in_sizes; (void)n_in; (void)out_size; (void)ws_size;
  const float* x  = (const float*)d_in[0];
  const float* cs = (const float*)d_in[1];
  const float* sn = (const float*)d_in[2];
  const float* Wq = (const float*)d_in[3];
  const float* Wk = (const float*)d_in[4];
  const float* Wv = (const float*)d_in[5];
  const float* Wo = (const float*)d_in[6];
  float* out = (float*)d_out;

  const size_t nX = (size_t)BB * SS * DM;   // 8,388,608 elems
  const size_t nW = (size_t)DM * DM;        // 4,194,304 elems

  _Float16* base = (_Float16*)d_ws;         // ~112 MB f16 scratch
  _Float16* xh   = base;
  _Float16* wqh  = xh + nX;
  _Float16* wkh  = wqh + nW;
  _Float16* wvh  = wkh + nW;
  _Float16* woh  = wvh + nW;
  _Float16* qlin = woh + nW;
  _Float16* klin = qlin + nX;
  _Float16* vtr  = klin + nX;
  _Float16* olin = vtr + nX;

  cvt_f32_f16<<<(unsigned)((nX + 255) / 256), 256, 0, stream>>>(x,  xh,  nX);
  cvt_f32_f16<<<(unsigned)((nW + 255) / 256), 256, 0, stream>>>(Wq, wqh, nW);
  cvt_f32_f16<<<(unsigned)((nW + 255) / 256), 256, 0, stream>>>(Wk, wkh, nW);
  cvt_f32_f16<<<(unsigned)((nW + 255) / 256), 256, 0, stream>>>(Wv, wvh, nW);
  cvt_f32_f16<<<(unsigned)((nW + 255) / 256), 256, 0, stream>>>(Wo, woh, nW);

  dim3 gg(BB * SS / 64, DM / 64);  // 64 x 32 workgroups, 4 waves each
  gemm_tdm_64x64<<<gg, 128, 0, stream>>>(xh, wqh, qlin, nullptr, BB * SS, DM, 0);
  gemm_tdm_64x64<<<gg, 128, 0, stream>>>(xh, wkh, klin, nullptr, BB * SS, DM, 0);
  gemm_tdm_64x64<<<gg, 128, 0, stream>>>(xh, wvh, vtr,  nullptr, BB * SS, DM, 2);

  const size_t nR = (size_t)BB * SS * NH * 64;
  rope_kernel<<<(unsigned)(nR / 256), 256, 0, stream>>>(qlin, cs, sn);
  rope_kernel<<<(unsigned)(nR / 256), 256, 0, stream>>>(klin, cs, sn);

  flash_attn<<<BB * NH * (SS / 16), 32, 0, stream>>>(qlin, klin, vtr, olin);

  gemm_tdm_64x64<<<gg, 128, 0, stream>>>(olin, woh, nullptr, out, BB * SS, DM, 1);
}
